// RQVAE_3264175145091
// MI455X (gfx1250) — matrix operand order repaired and checked by reading, not compiled, not run on hardware
//
#include <hip/hip_runtime.h>

typedef __attribute__((ext_vector_type(16))) _Float16 v16h;
typedef __attribute__((ext_vector_type(8)))  _Float16 v8h;
typedef __attribute__((ext_vector_type(8)))  float    v8f;

#define LDH 520          // half-stride of LDS activation buffers (1040 B, 16B aligned)
#define BM  64           // batch rows per workgroup

// ---------------------------------------------------------------------------
// Fused layer GEMM: C[64 x N] = relu?(A[64 x K] * W^T + b)
// SRC: 0 = A from LDS f16 buffer, 1 = A from global f32 x
// DST: 0 = LDS f16 + ReLU, 1 = f32 zbuf+zorig (encoder latent), 2 = global f32
// ---------------------------------------------------------------------------
template<int SRC, int DST>
__device__ __forceinline__ void layer_gemm(
    const _Float16* __restrict__ srcL, const float* __restrict__ srcG,
    const _Float16* __restrict__ wt,   const float* __restrict__ bias,
    _Float16* __restrict__ dstL, float* __restrict__ dstG,
    float* __restrict__ zb, float* __restrict__ zo,
    int K, int N, int row0, int wave, int lane)
{
    const int ntiles = N >> 4;
    const int total  = ntiles << 2;          // (64/16) row tiles * ntiles
    const int lcol   = lane & 15;
    const int khalf  = lane >> 4;

    for (int t = wave; t < total; t += 8) {
        const int tm  = (t / ntiles) << 4;
        const int tn  = (t % ntiles) << 4;
        const int col = tn + lcol;

        __builtin_prefetch(wt + (size_t)col * K, 0, 1);   // global_prefetch_b8

        v8f acc = {};
        for (int kb = 0; kb < K; kb += 32) {
            // ---- A fragment (16x32 f16, ISA layout) ----
            v16h a;
            if (SRC == 0) {
                const _Float16* p = srcL + (tm + lcol) * LDH + kb + khalf * 8;
                v8h lo = *(const v8h*)p;          // K 0-7   (or 8-15)
                v8h hi = *(const v8h*)(p + 16);   // K 16-23 (or 24-31)
#pragma unroll
                for (int i = 0; i < 8; ++i) { a[i] = lo[i]; a[8 + i] = hi[i]; }
            } else {
                const float* p = srcG + (size_t)(row0 + tm + lcol) * K + kb + khalf * 8;
#pragma unroll
                for (int i = 0; i < 8; ++i) {
                    a[i]     = (_Float16)p[i];
                    a[8 + i] = (_Float16)p[16 + i];
                }
            }
            // ---- B fragment (32x16 f16): Wt[col][kb + khalf*8 .. +7], [kb + 16 + khalf*8 .. +7] ----
            const _Float16* q = wt + (size_t)col * K + kb + khalf * 8;
            v8h blo = *(const v8h*)q;
            v8h bhi = *(const v8h*)(q + 16);
            v16h b;
#pragma unroll
            for (int i = 0; i < 8; ++i) { b[i] = blo[i]; b[8 + i] = bhi[i]; }

            acc = __builtin_amdgcn_wmma_f32_16x16x32_f16(
                      false, a, false, b, (short)0, acc, false, false);
        }

        const float bv    = bias[col];
        const int   rbase = tm + khalf * 8;
#pragma unroll
        for (int i = 0; i < 8; ++i) {
            float v = acc[i] + bv;
            const int row = rbase + i;
            if (DST == 0) {
                v = v > 0.f ? v : 0.f;
                dstL[row * LDH + col] = (_Float16)v;
            } else if (DST == 1) {
                zb[row * 32 + col] = v;
                zo[row * 32 + col] = v;
            } else {
                dstG[(size_t)(row0 + row) * 768 + col] = v;
            }
        }
    }
}

// ---------------------------------------------------------------------------
// Fused RQ-VAE forward: enc MLP -> 3-level RVQ -> dec MLP, one 64-row slab/WG
// ---------------------------------------------------------------------------
__global__ __launch_bounds__(256) void rqvae_fused(
    const float* __restrict__ x,
    const _Float16* __restrict__ we0, const _Float16* __restrict__ we1,
    const _Float16* __restrict__ we2, const _Float16* __restrict__ we3,
    const float* __restrict__ be0, const float* __restrict__ be1,
    const float* __restrict__ be2, const float* __restrict__ be3,
    const _Float16* __restrict__ wd0, const _Float16* __restrict__ wd1,
    const _Float16* __restrict__ wd2, const _Float16* __restrict__ wd3,
    const float* __restrict__ bd0, const float* __restrict__ bd1,
    const float* __restrict__ bd2, const float* __restrict__ bd3,
    const float* __restrict__ codebooks,
    float* __restrict__ out_dec, float* __restrict__ out_r,
    float* __restrict__ out_e,   float* __restrict__ out_counts,
    float* __restrict__ out_q)
{
    __shared__ __align__(16) unsigned char smem[152832];
    _Float16* bufA  = (_Float16*)(smem);            // 64*520 halfs = 66560 B
    _Float16* bufB  = (_Float16*)(smem + 66560);    // 66560 B
    float*    zbuf  = (float*)(smem + 133120);      // 64*32 f32 residual
    float*    zorig = (float*)(smem + 141312);      // 64*32 f32 original z
    float*    cnorm = (float*)(smem + 149504);      // 256 f32
    float*    red_d = (float*)(smem + 150528);      // 64*4 f32
    int*      red_i = (int*)  (smem + 151552);      // 64*4 i32
    int*      idxb  = (int*)  (smem + 152576);      // 64 i32
    float*    cbuf  = (float*)bufB;                 // 256*32 f32, aliased (free then)

    const int tid  = threadIdx.x;
    const int wave = tid >> 5;
    const int lane = tid & 31;
    const int row0 = blockIdx.x * BM;

    // ---------------- encoder ----------------
    layer_gemm<1,0>(nullptr, x,   we0, be0, bufA, nullptr, nullptr, nullptr, 768, 512, row0, wave, lane);
    __syncthreads();
    layer_gemm<0,0>(bufA, nullptr, we1, be1, bufB, nullptr, nullptr, nullptr, 512, 256, row0, wave, lane);
    __syncthreads();
    layer_gemm<0,0>(bufB, nullptr, we2, be2, bufA, nullptr, nullptr, nullptr, 256, 128, row0, wave, lane);
    __syncthreads();
    layer_gemm<0,1>(bufA, nullptr, we3, be3, nullptr, nullptr, zbuf, zorig,  128,  32, row0, wave, lane);
    __syncthreads();

    // ---------------- residual vector quantization ----------------
    const unsigned cbuf_lds = (unsigned)(uintptr_t)cbuf;  // generic LDS addr[31:0]
    for (int l = 0; l < 3; ++l) {
        // stage codebook level l: 32 KB global -> LDS on the async DMA path
        // (GLOBAL_LOAD_ASYNC_TO_LDS_B128, tracked by ASYNCcnt, bypasses VGPRs)
        {
            const float* cb = codebooks + (size_t)l * 8192;
#pragma unroll
            for (int it = 0; it < 8; ++it) {
                const int      elt  = (it * 256 + tid) * 4;     // 4 floats / lane
                const unsigned ldsa = cbuf_lds + (unsigned)(elt * 4);
                const float*   g    = cb + elt;
                asm volatile("global_load_async_to_lds_b128 %0, %1, off"
                             :: "v"(ldsa), "v"(g) : "memory");
            }
            asm volatile("s_wait_asynccnt 0x0" ::: "memory");
        }
        __syncthreads();
        {   // per-code squared norm
            float s = 0.f;
#pragma unroll
            for (int d = 0; d < 32; ++d) { float c = cbuf[tid * 32 + d]; s += c * c; }
            cnorm[tid] = s;
        }
        __syncthreads();
        {   // argmin over |c|^2 - 2 r.c ; 4 threads per row, 64 codes each
            const int row = tid & 63, chunk = tid >> 6;
            float best = 3.4e38f; int bi = chunk * 64;
            for (int c = chunk * 64; c < chunk * 64 + 64; ++c) {
                float dot = 0.f;
#pragma unroll
                for (int d = 0; d < 32; ++d) dot += zbuf[row * 32 + d] * cbuf[c * 32 + d];
                const float s = cnorm[c] - 2.f * dot;
                if (s < best) { best = s; bi = c; }
            }
            red_d[row * 4 + chunk] = best;
            red_i[row * 4 + chunk] = bi;
        }
        __syncthreads();
        if (tid < 64) {   // reduce 4 chunks (ascending -> first-index tie break)
            float bd = red_d[tid * 4]; int bi = red_i[tid * 4];
#pragma unroll
            for (int c = 1; c < 4; ++c) {
                const float v = red_d[tid * 4 + c];
                if (v < bd) { bd = v; bi = red_i[tid * 4 + c]; }
            }
            idxb[tid] = bi;
            out_q[(size_t)(row0 + tid) * 3 + l] = (float)bi;
            atomicAdd(&out_counts[l * 256 + bi], 1.0f);
        }
        __syncthreads();
        {   // emit r_l, e_l; update residual
            const int row = tid >> 2;
            const int d0  = (tid & 3) * 8;
            const int id  = idxb[row];
            const size_t ob = ((size_t)(row0 + row) * 3 + l) * 32 + d0;
#pragma unroll
            for (int j = 0; j < 8; ++j) {
                const float rv = zbuf[row * 32 + d0 + j];
                const float ev = cbuf[id * 32 + d0 + j];
                out_r[ob + j] = rv;
                out_e[ob + j] = ev;
                zbuf[row * 32 + d0 + j] = rv - ev;
            }
        }
        __syncthreads();
    }

    // decoder input: z_hat (fwd value) = sum_e = z - r_final
    {
        const int row = tid >> 2;
        const int d0  = (tid & 3) * 8;
#pragma unroll
        for (int j = 0; j < 8; ++j)
            bufA[row * LDH + d0 + j] =
                (_Float16)(zorig[row * 32 + d0 + j] - zbuf[row * 32 + d0 + j]);
    }
    __syncthreads();

    // ---------------- decoder ----------------
    layer_gemm<0,0>(bufA, nullptr, wd0, bd0, bufB, nullptr, nullptr, nullptr,  32, 128, row0, wave, lane);
    __syncthreads();
    layer_gemm<0,0>(bufB, nullptr, wd1, bd1, bufA, nullptr, nullptr, nullptr, 128, 256, row0, wave, lane);
    __syncthreads();
    layer_gemm<0,0>(bufA, nullptr, wd2, bd2, bufB, nullptr, nullptr, nullptr, 256, 512, row0, wave, lane);
    __syncthreads();
    layer_gemm<0,2>(bufB, nullptr, wd3, bd3, nullptr, out_dec, nullptr, nullptr, 512, 768, row0, wave, lane);
}

// ---------------------------------------------------------------------------
// W [K][N] f32  ->  Wt [N][K] f16   (B-operand friendly layout)
// ---------------------------------------------------------------------------
__global__ void wt_convert(const float* __restrict__ W, _Float16* __restrict__ Wt,
                           int K, int N)
{
    const int i = blockIdx.x * blockDim.x + threadIdx.x;
    if (i >= K * N) return;
    const int n = i / K, k = i - n * K;
    Wt[i] = (_Float16)W[(size_t)k * N + n];
}

__global__ void zero_counts(float* __restrict__ c)
{
    const int i = blockIdx.x * blockDim.x + threadIdx.x;
    if (i < 768) c[i] = 0.f;
}

// ---------------------------------------------------------------------------
extern "C" void kernel_launch(void* const* d_in, const int* in_sizes, int n_in,
                              void* d_out, int out_size, void* d_ws, size_t ws_size,
                              hipStream_t stream)
{
    (void)in_sizes; (void)n_in; (void)out_size; (void)ws_size;
    // dict order: x, {enc_W i, enc_b i, dec_W i, dec_b i} x4, codebooks
    const float* x = (const float*)d_in[0];
    const float* eW[4] = {(const float*)d_in[1],  (const float*)d_in[5],
                          (const float*)d_in[9],  (const float*)d_in[13]};
    const float* eB[4] = {(const float*)d_in[2],  (const float*)d_in[6],
                          (const float*)d_in[10], (const float*)d_in[14]};
    const float* dW[4] = {(const float*)d_in[3],  (const float*)d_in[7],
                          (const float*)d_in[11], (const float*)d_in[15]};
    const float* dB[4] = {(const float*)d_in[4],  (const float*)d_in[8],
                          (const float*)d_in[12], (const float*)d_in[16]};
    const float* cbk = (const float*)d_in[17];

    const int encD[5] = {768, 512, 256, 128, 32};
    const int decD[5] = {32, 128, 256, 512, 768};

    _Float16* ws = (_Float16*)d_ws;
    _Float16 *we[4], *wd[4];
    size_t off = 0;
    for (int i = 0; i < 4; ++i) { we[i] = ws + off; off += (size_t)encD[i] * encD[i + 1]; }
    for (int i = 0; i < 4; ++i) { wd[i] = ws + off; off += (size_t)decD[i] * decD[i + 1]; }

    for (int i = 0; i < 4; ++i) {
        int n = encD[i] * encD[i + 1];
        wt_convert<<<(n + 255) / 256, 256, 0, stream>>>(eW[i], we[i], encD[i], encD[i + 1]);
        n = decD[i] * decD[i + 1];
        wt_convert<<<(n + 255) / 256, 256, 0, stream>>>(dW[i], wd[i], decD[i], decD[i + 1]);
    }

    float* out     = (float*)d_out;                  // concatenated tuple, f32
    float* out_dec = out;                            // [131072, 768]
    float* out_r   = out_dec + (size_t)131072 * 768; // [131072, 3, 32]
    float* out_e   = out_r   + (size_t)131072 * 96;  // [131072, 3, 32]
    float* out_cnt = out_e   + (size_t)131072 * 96;  // [3, 256]
    float* out_q   = out_cnt + 768;                  // [131072, 3]

    zero_counts<<<3, 256, 0, stream>>>(out_cnt);

    rqvae_fused<<<131072 / BM, 256, 0, stream>>>(
        x,
        we[0], we[1], we[2], we[3], eB[0], eB[1], eB[2], eB[3],
        wd[0], wd[1], wd[2], wd[3], dB[0], dB[1], dB[2], dB[3],
        cbk, out_dec, out_r, out_e, out_cnt, out_q);
}
